// GraphAutoEncoderCora_3504693313768
// MI455X (gfx1250) — matrix operand order repaired and checked by reading, not compiled, not run on hardware
//
#include <hip/hip_runtime.h>

// ---------------------------------------------------------------------------
// GraphAutoEncoderCora on gfx1250 (MI455X): bf16 WMMA GEMMs + L2-atomic SpMM.
// Decoder z@z^T (38.6 GFLOP, 604MB store) dominates -> v_wmma_f32_16x16x32_bf16
// with 32x32 register tile per wave, operands read directly row-major (B^T==z).
// adj_recon stores use the NT temporal hint so the 604MB stream (3x the 192MB
// L2) does not evict the L2-resident z operand.
// ---------------------------------------------------------------------------

typedef __attribute__((ext_vector_type(16))) __bf16       v16bf;
typedef __attribute__((ext_vector_type(8)))  float        v8f;
typedef __attribute__((ext_vector_type(4)))  unsigned int u32x4;

union Frag {
  v16bf v;
  u32x4 q[2];
};
static_assert(sizeof(Frag) == 32, "frag must be 8 VGPRs");

__device__ __forceinline__ unsigned short f32_to_bf16_rne(float f) {
  union { float f; unsigned int u; } c; c.f = f;
  unsigned int u = c.u;
  u += 0x7FFFu + ((u >> 16) & 1u);   // round-to-nearest-even
  return (unsigned short)(u >> 16);
}

__device__ __forceinline__ float sigmoidf_(float x) {
  return 1.0f / (1.0f + __expf(-x));
}

__device__ __forceinline__ v8f wmma_bf16(const Frag& a, const Frag& b, v8f c) {
  // D = A(16x32 bf16) * B(32x16 bf16) + C(16x16 f32)
  return __builtin_amdgcn_wmma_f32_16x16x32_bf16(
      /*neg_a=*/false, a.v, /*neg_b=*/false, b.v,
      /*c_mod=*/(short)0, c, /*reuse_a=*/false, /*reuse_b=*/false);
}

// ---------------------------------------------------------------------------
// Elementwise helpers
// ---------------------------------------------------------------------------
__global__ void k_cvt_bf16(const float* __restrict__ in,
                           unsigned short* __restrict__ out, int n) {
  int i = blockIdx.x * blockDim.x + threadIdx.x;
  if (i < n) out[i] = f32_to_bf16_rne(in[i]);
}

__global__ void k_relu_cvt_bf16(const float* __restrict__ in,
                                unsigned short* __restrict__ out, int n) {
  int i = blockIdx.x * blockDim.x + threadIdx.x;
  if (i < n) {
    float v = in[i];
    out[i] = f32_to_bf16_rne(v > 0.0f ? v : 0.0f);
  }
}

__global__ void k_zero_f32(float* __restrict__ p, int n) {
  int i = blockIdx.x * blockDim.x + threadIdx.x;
  if (i < n) p[i] = 0.0f;
}

// W [K,Nc] f32 (row-major) -> WT [Nc,K] bf16 (row-major), i.e. B^T layout
__global__ void k_transpose_cvt_bf16(const float* __restrict__ W,
                                     unsigned short* __restrict__ WT,
                                     int K, int Nc) {
  int i = blockIdx.x * blockDim.x + threadIdx.x;
  if (i >= K * Nc) return;
  int k = i / Nc, n = i - k * Nc;
  WT[(size_t)n * K + k] = f32_to_bf16_rne(W[i]);
}

// ---------------------------------------------------------------------------
// Generic GEMM: C[M,Nc] f32 = A[M,K] bf16 (row-major) x B, with BT[Nc,K] bf16.
// One 16x16 tile per wave; K multiple of 32; M,Nc multiples of 16.
// A-frag: lane L -> row m=L%16, 16B chunks at k0+half*8 and k0+16+half*8.
// B-frag: lane L -> col n=L%16, 16B chunks at k0+half*8 and k0+16+half*8 of BT row n.
// ---------------------------------------------------------------------------
__global__ void k_gemm_bf16(const unsigned short* __restrict__ A,
                            const unsigned short* __restrict__ BT,
                            float* __restrict__ C,
                            int M, int Nc, int K) {
  int wave = (blockIdx.x * blockDim.x + threadIdx.x) >> 5;
  int lane = threadIdx.x & 31;
  int tiles_n = Nc >> 4;
  int tiles_m = M >> 4;
  int tm = wave / tiles_n;
  int tn = wave - tm * tiles_n;
  if (tm >= tiles_m) return;                // uniform per wave -> EXEC all-ones
  int half = lane >> 4;
  int l16  = lane & 15;

  const unsigned short* Arow = A  + (size_t)(tm * 16 + l16) * K;
  const unsigned short* Brow = BT + (size_t)(tn * 16 + l16) * K;

  v8f acc = {};
  for (int k0 = 0; k0 < K; k0 += 32) {
    Frag a, b;
    a.q[0] = *(const u32x4*)(Arow + k0 + half * 8);
    a.q[1] = *(const u32x4*)(Arow + k0 + 16 + half * 8);
    b.q[0] = *(const u32x4*)(Brow + k0 + half * 8);
    b.q[1] = *(const u32x4*)(Brow + k0 + 16 + half * 8);
    acc = wmma_bf16(a, b, acc);
  }

  float* Crow = C + (size_t)(tm * 16 + half * 8) * Nc + (tn * 16 + l16);
#pragma unroll
  for (int r = 0; r < 8; ++r) Crow[(size_t)r * Nc] = acc[r];
}

// ---------------------------------------------------------------------------
// Decoder: out = sigmoid(Z @ Z^T), Z [Nn,K] bf16. B^T of Z^T is Z itself, so
// both operands stream from the same row-major buffer. 32x32 tile per wave
// (2x2 WMMA register tile) for 2x operand reuse; Z (3MB) is L2-resident.
// Output stores are non-temporal: 604MB > L2, never re-read.
// ---------------------------------------------------------------------------
__global__ void k_decoder_sigmoid(const unsigned short* __restrict__ Z,
                                  float* __restrict__ out,
                                  int Nn, int K) {
  int wave = (blockIdx.x * blockDim.x + threadIdx.x) >> 5;
  int lane = threadIdx.x & 31;
  int tiles = Nn >> 5;
  int tm = wave / tiles;
  int tn = wave - tm * tiles;
  if (tm >= tiles) return;
  int half = lane >> 4;
  int l16  = lane & 15;

  const unsigned short* A0 = Z + (size_t)(tm * 32 + l16) * K;
  const unsigned short* A1 = A0 + (size_t)16 * K;
  const unsigned short* B0 = Z + (size_t)(tn * 32 + l16) * K;
  const unsigned short* B1 = B0 + (size_t)16 * K;

  v8f c00 = {}, c01 = {}, c10 = {}, c11 = {};
  for (int k0 = 0; k0 < K; k0 += 32) {
    Frag a0, a1, b0, b1;
    a0.q[0] = *(const u32x4*)(A0 + k0 + half * 8);
    a0.q[1] = *(const u32x4*)(A0 + k0 + 16 + half * 8);
    a1.q[0] = *(const u32x4*)(A1 + k0 + half * 8);
    a1.q[1] = *(const u32x4*)(A1 + k0 + 16 + half * 8);
    b0.q[0] = *(const u32x4*)(B0 + k0 + half * 8);
    b0.q[1] = *(const u32x4*)(B0 + k0 + 16 + half * 8);
    b1.q[0] = *(const u32x4*)(B1 + k0 + half * 8);
    b1.q[1] = *(const u32x4*)(B1 + k0 + 16 + half * 8);
    c00 = wmma_bf16(a0, b0, c00);
    c01 = wmma_bf16(a0, b1, c01);
    c10 = wmma_bf16(a1, b0, c10);
    c11 = wmma_bf16(a1, b1, c11);
  }

  int m0 = tm * 32 + half * 8;
  int n0 = tn * 32 + l16;
#pragma unroll
  for (int r = 0; r < 8; ++r) {
    size_t ro  = (size_t)(m0 + r) * Nn;
    size_t ro2 = (size_t)(m0 + 16 + r) * Nn;
    __builtin_nontemporal_store(sigmoidf_(c00[r]), out + ro  + n0);
    __builtin_nontemporal_store(sigmoidf_(c01[r]), out + ro  + n0 + 16);
    __builtin_nontemporal_store(sigmoidf_(c10[r]), out + ro2 + n0);
    __builtin_nontemporal_store(sigmoidf_(c11[r]), out + ro2 + n0 + 16);
  }
}

// ---------------------------------------------------------------------------
// SpMM (COO segment-sum): out[src[e]] += val[e] * dense[dst[e]].
// One wave per edge: coalesced row gather (dense rows are L2-resident) and
// no-return global_atomic_add_f32 scatter (STOREcnt, fire-and-forget).
// ---------------------------------------------------------------------------
__global__ void k_spmm_atomic(const int* __restrict__ src, const int* __restrict__ dst,
                              const float* __restrict__ val,
                              const float* __restrict__ dense,
                              float* __restrict__ out, int E, int H) {
  int t = blockIdx.x * blockDim.x + threadIdx.x;
  int e = t >> 5, lane = t & 31;
  if (e >= E) return;
  int s = src[e], d = dst[e];
  float v = val[e];
  const float* drow = dense + (size_t)d * H;
  float*       orow = out   + (size_t)s * H;
  for (int f = lane; f < H; f += 32)
    atomicAdd(orow + f, v * drow[f]);
}

// Fused dual SpMM for mu/logvar (shared indices -> half the index traffic)
__global__ void k_spmm2_atomic(const int* __restrict__ src, const int* __restrict__ dst,
                               const float* __restrict__ val,
                               const float* __restrict__ d1, const float* __restrict__ d2,
                               float* __restrict__ o1, float* __restrict__ o2,
                               int E, int H) {
  int t = blockIdx.x * blockDim.x + threadIdx.x;
  int e = t >> 5, lane = t & 31;
  if (e >= E) return;
  int s = src[e], d = dst[e];
  float v = val[e];
  const float* r1 = d1 + (size_t)d * H;
  const float* r2 = d2 + (size_t)d * H;
  float* w1 = o1 + (size_t)s * H;
  float* w2 = o2 + (size_t)s * H;
  for (int f = lane; f < H; f += 32) {
    atomicAdd(w1 + f, v * r1[f]);
    atomicAdd(w2 + f, v * r2[f]);
  }
}

// ---------------------------------------------------------------------------
// Host orchestration
// ---------------------------------------------------------------------------
extern "C" void kernel_launch(void* const* d_in, const int* in_sizes, int n_in,
                              void* d_out, int out_size, void* d_ws, size_t ws_size,
                              hipStream_t stream) {
  constexpr int N = 12288, F = 512, H1 = 256, H2 = 128, E = 393216;
  (void)in_sizes; (void)n_in; (void)out_size; (void)ws_size;

  const float* x   = (const float*)d_in[0];
  const int*   src = (const int*)d_in[1];
  const int*   dst = (const int*)d_in[2];
  const float* ev  = (const float*)d_in[3];
  const float* W1  = (const float*)d_in[4];
  const float* W2  = (const float*)d_in[5];
  const float* W3  = (const float*)d_in[6];

  float* out    = (float*)d_out;               // adj_recon [N,N]
  float* mu     = out + (size_t)N * N;         // [N,H2]
  float* logvar = mu + (size_t)N * H2;         // [N,H2]

  // Workspace carve-out (256B aligned), ~60 MB total.
  char* ws = (char*)d_ws;
  size_t off = 0;
  auto take = [&](size_t bytes) -> char* {
    char* p = ws + off;
    off = (off + bytes + 255) & ~(size_t)255;
    return p;
  };
  unsigned short* xb   = (unsigned short*)take((size_t)N * F * 2);   // x bf16
  unsigned short* w1t  = (unsigned short*)take((size_t)F * H1 * 2);  // W1^T bf16
  unsigned short* w2t  = (unsigned short*)take((size_t)H1 * H2 * 2); // W2^T bf16
  unsigned short* w3t  = (unsigned short*)take((size_t)H1 * H2 * 2); // W3^T bf16
  float*          sup1 = (float*)take((size_t)N * H1 * 4);           // x@W1
  float*          h1a  = (float*)take((size_t)N * H1 * 4);           // spmm accum
  unsigned short* h1b  = (unsigned short*)take((size_t)N * H1 * 2);  // relu(h1) bf16
  float*          sup2 = (float*)take((size_t)N * H2 * 4);           // h1@W2
  float*          sup3 = (float*)take((size_t)N * H2 * 4);           // h1@W3
  unsigned short* mub  = (unsigned short*)take((size_t)N * H2 * 2);  // z bf16

  const int T = 256;   // 8 waves per block

  // 1) operand precision conversion / weight transposition
  k_cvt_bf16<<<(N * F + T - 1) / T, T, 0, stream>>>(x, xb, N * F);
  k_transpose_cvt_bf16<<<(F * H1 + T - 1) / T, T, 0, stream>>>(W1, w1t, F, H1);
  k_transpose_cvt_bf16<<<(H1 * H2 + T - 1) / T, T, 0, stream>>>(W2, w2t, H1, H2);
  k_transpose_cvt_bf16<<<(H1 * H2 + T - 1) / T, T, 0, stream>>>(W3, w3t, H1, H2);

  // 2) sup1 = x @ W1   (WMMA, M=12288 Nc=256 K=512)
  {
    int waves = (N / 16) * (H1 / 16);
    k_gemm_bf16<<<(waves + 7) / 8, T, 0, stream>>>(xb, w1t, sup1, N, H1, F);
  }

  // 3) h1 = relu(spmm(sup1))
  k_zero_f32<<<(N * H1 + T - 1) / T, T, 0, stream>>>(h1a, N * H1);
  k_spmm_atomic<<<(E * 32 + T - 1) / T, T, 0, stream>>>(src, dst, ev, sup1, h1a, E, H1);
  k_relu_cvt_bf16<<<(N * H1 + T - 1) / T, T, 0, stream>>>(h1a, h1b, N * H1);

  // 4) sup2 = h1@W2, sup3 = h1@W3   (M=12288 Nc=128 K=256)
  {
    int waves = (N / 16) * (H2 / 16);
    k_gemm_bf16<<<(waves + 7) / 8, T, 0, stream>>>(h1b, w2t, sup2, N, H2, H1);
    k_gemm_bf16<<<(waves + 7) / 8, T, 0, stream>>>(h1b, w3t, sup3, N, H2, H1);
  }

  // 5) mu = spmm(sup2), logvar = spmm(sup3)  (mu/logvar contiguous in d_out)
  k_zero_f32<<<(2 * N * H2 + T - 1) / T, T, 0, stream>>>(mu, 2 * N * H2);
  k_spmm2_atomic<<<(E * 32 + T - 1) / T, T, 0, stream>>>(src, dst, ev, sup2, sup3,
                                                         mu, logvar, E, H2);

  // 6) z = mu; adj = sigmoid(z @ z^T)   (dominant GEMM: 12288x12288x128)
  k_cvt_bf16<<<(N * H2 + T - 1) / T, T, 0, stream>>>(mu, mub, N * H2);
  {
    int waves = (N / 32) * (N / 32);   // 147456 waves, 32x32 tile each
    k_decoder_sigmoid<<<(waves + 7) / 8, T, 0, stream>>>(mub, out, N, H2);
  }
}
